// MemoryBank_49460843380950
// MI455X (gfx1250) — matrix operand order repaired and checked by reading, not compiled, not run on hardware
//
#include <hip/hip_runtime.h>

// ---------------------------------------------------------------------------
// MemoryBank retrieval: sim = Q @ M^T (bf16 WMMA, f32 accum), per-query top-5,
// gather rows.  B=1024, K=262144, D=64, topk=5.
// ---------------------------------------------------------------------------

#define B_Q     1024
#define K_MEM   262144
#define D_DIM   64
#define TOPK    5

#define NCHUNK  64                      // K chunks (grid.y of scoring kernel)
#define CHUNK   (K_MEM / NCHUNK)        // 4096 rows per chunk
#define ROWS    128                     // rows staged in LDS per iteration (16KB)
#define QSUPER  64                      // queries per block (4 waves x 16)
#define NQT     (B_Q / QSUPER)          // 16 query supertiles

typedef __attribute__((ext_vector_type(16))) __bf16 v16bf;
typedef __attribute__((ext_vector_type(8)))  float  v8f;
typedef __attribute__((ext_vector_type(4)))  int    v4i_t;

// ---- optional gfx1250 async global->LDS staging path ----------------------
#if defined(__has_builtin)
#if __has_builtin(__builtin_amdgcn_global_load_async_to_lds_b128)
#define HAVE_ASYNC_LDS 1
#endif
#endif
#ifndef HAVE_ASYNC_LDS
#define HAVE_ASYNC_LDS 0
#endif

#if HAVE_ASYNC_LDS
typedef __attribute__((address_space(1))) v4i_t as1_v4i;
typedef __attribute__((address_space(3))) v4i_t as3_v4i;
#if __has_builtin(__builtin_amdgcn_s_wait_asynccnt)
#define ASYNC_WAIT0() __builtin_amdgcn_s_wait_asynccnt(0)
#else
#define ASYNC_WAIT0() asm volatile("s_wait_asynccnt 0x0" ::: "memory")
#endif

// Immediate offset must be a constant expression -> template parameter.
template <int OFF>
__device__ __forceinline__ void async_cp_b128(const unsigned short* src,
                                              unsigned short* dst) {
    __builtin_amdgcn_global_load_async_to_lds_b128(
        (as1_v4i*)src, (as3_v4i*)dst, OFF, 0);
}
#endif

union Frag {                            // 16 bf16 = 32B = 2 x uint4
    v16bf v;
    uint4 u[2];
};

__device__ __forceinline__ unsigned short f32_to_bf16(float f) {
    unsigned int u = __float_as_uint(f);
    unsigned int r = (u + 0x7FFFu + ((u >> 16) & 1u)) >> 16;   // RNE
    return (unsigned short)r;
}

// ---------------------------------------------------------------------------
// Kernel 0: f32 -> bf16 conversion (vectorized: float4 in, uint2 out)
// ---------------------------------------------------------------------------
__global__ void cvt_bf16_kernel(const float* __restrict__ in,
                                unsigned short* __restrict__ out, int n4) {
    int i = blockIdx.x * blockDim.x + threadIdx.x;
    if (i >= n4) return;
    float4 f = ((const float4*)in)[i];
    uint2 p;
    p.x = (unsigned)f32_to_bf16(f.x) | ((unsigned)f32_to_bf16(f.y) << 16);
    p.y = (unsigned)f32_to_bf16(f.z) | ((unsigned)f32_to_bf16(f.w) << 16);
    ((uint2*)out)[i] = p;
}

// ---------------------------------------------------------------------------
// Kernel 1: score a 64-query supertile against a 4096-row K-chunk.
//   grid = (NQT, NCHUNK), block = 128 (4 waves; each wave owns 16 queries).
//   Memory rows staged via LDS (16KB tiles) and shared by the 4 waves.
//   Emits per-(query, chunk) top-5 (val, idx) candidates.
// ---------------------------------------------------------------------------
__global__ __launch_bounds__(128, 1)
void score_topk_kernel(const unsigned short* __restrict__ Mbf,   // [K][64]
                       const unsigned short* __restrict__ Qbf,   // [B][64]
                       float* __restrict__ cand_val,             // [B][NCHUNK][5]
                       int*   __restrict__ cand_idx) {
    __shared__ unsigned short ldsM[ROWS * D_DIM];          // 16 KB staging
    __shared__ float lds_val[QSUPER][16][TOPK];            // 20 KB
    __shared__ int   lds_idx[QSUPER][16][TOPK];            // 20 KB

    const int tid  = threadIdx.x;
    const int wave = tid >> 5;
    const int lane = tid & 31;
    const int col  = lane & 15;         // N / M row within 16-tile
    const int half = lane >> 4;

    const int qbase = blockIdx.x * QSUPER + wave * 16;
    const int rbase = blockIdx.y * CHUNK;

    // --- A fragments for this wave's 16 queries (held in registers) --------
    // elems 0-7 : K in [8*half, 8*half+8);  elems 8-15 : K in [16+8*half, +8)
    Frag a0, a1;
    {
        const unsigned short* qrow = Qbf + (size_t)(qbase + col) * D_DIM;
        a0.u[0] = *(const uint4*)(qrow + 8 * half);
        a0.u[1] = *(const uint4*)(qrow + 8 * half + 16);
        a1.u[0] = *(const uint4*)(qrow + 32 + 8 * half);
        a1.u[1] = *(const uint4*)(qrow + 32 + 8 * half + 16);
    }

    // --- per-lane sorted top-5 for 8 queries (j -> query qbase+8*half+j) ---
    float tv[8][TOPK];
    int   ti[8][TOPK];
#pragma unroll
    for (int j = 0; j < 8; ++j)
#pragma unroll
        for (int t = 0; t < TOPK; ++t) { tv[j][t] = -3.0e38f; ti[j][t] = 0; }

    // --- stream the K chunk through LDS ------------------------------------
    for (int kt = 0; kt < CHUNK / ROWS; ++kt) {
        const int row0 = rbase + kt * ROWS;

        __syncthreads();                               // prior tile reads done
#if HAVE_ASYNC_LDS
        {
            // 128 rows * 128B = 16KB = 128 thr * 8 async b128 (no VGPR transit)
            const unsigned short* src = Mbf + (size_t)row0 * D_DIM + (size_t)tid * 8;
            unsigned short* dstp = ldsM + tid * 8;
            async_cp_b128<0>(src, dstp);
            async_cp_b128<2048>(src, dstp);
            async_cp_b128<4096>(src, dstp);
            async_cp_b128<6144>(src, dstp);
            async_cp_b128<8192>(src, dstp);
            async_cp_b128<10240>(src, dstp);
            async_cp_b128<12288>(src, dstp);
            async_cp_b128<14336>(src, dstp);
            ASYNC_WAIT0();                             // own writes in LDS
        }
#else
        {
            const uint4* src = (const uint4*)(Mbf + (size_t)row0 * D_DIM);
            uint4* dst = (uint4*)ldsM;
#pragma unroll
            for (int i = 0; i < 8; ++i) dst[tid + i * 128] = src[tid + i * 128];
        }
#endif
        if (kt + 1 < CHUNK / ROWS)                     // gfx1250 prefetch path
            __builtin_prefetch((const void*)(Mbf + (size_t)(row0 + ROWS) * D_DIM
                                             + (size_t)tid * 64), 0, 1);
        __syncthreads();                               // staging visible to all

#pragma unroll
        for (int sub = 0; sub < ROWS / 16; ++sub) {
            // B fragment: elems 0-7 : K in [8*half, +8); elems 8-15 : K in [16+8*half, +8)
            const unsigned short* brow =
                ldsM + (sub * 16 + col) * D_DIM + 8 * half;
            Frag b0, b1;
            b0.u[0] = *(const uint4*)(brow);
            b0.u[1] = *(const uint4*)(brow + 16);
            b1.u[0] = *(const uint4*)(brow + 32);
            b1.u[1] = *(const uint4*)(brow + 48);

            v8f c = {};
            c = __builtin_amdgcn_wmma_f32_16x16x32_bf16(
                    false, a0.v, false, b0.v, (short)0, c, false, false);
            c = __builtin_amdgcn_wmma_f32_16x16x32_bf16(
                    false, a1.v, false, b1.v, (short)0, c, false, false);

            const int ridx = row0 + sub * 16 + col;    // this lane's mem row
#pragma unroll
            for (int j = 0; j < 8; ++j) {
                float val = c[j];
                if (val > tv[j][TOPK - 1]) {           // usually skipped
                    int id = ridx;
#pragma unroll
                    for (int t = 0; t < TOPK; ++t) {
                        if (val > tv[j][t]) {
                            float fv = tv[j][t]; tv[j][t] = val; val = fv;
                            int   fi = ti[j][t]; ti[j][t] = id;  id  = fi;
                        }
                    }
                }
            }
        }
    }

    // --- merge the 16 lane-columns per query, emit chunk candidates --------
    __syncthreads();
#pragma unroll
    for (int j = 0; j < 8; ++j) {
        int ql = wave * 16 + half * 8 + j;
#pragma unroll
        for (int t = 0; t < TOPK; ++t) {
            lds_val[ql][col][t] = tv[j][t];
            lds_idx[ql][col][t] = ti[j][t];
        }
    }
    __syncthreads();

    if (tid < QSUPER) {
        float bv[TOPK]; int bi[TOPK];
#pragma unroll
        for (int t = 0; t < TOPK; ++t) { bv[t] = -3.0e38f; bi[t] = 0; }
        for (int cc = 0; cc < 16; ++cc) {
#pragma unroll
            for (int s = 0; s < TOPK; ++s) {
                float val = lds_val[tid][cc][s];
                int   id  = lds_idx[tid][cc][s];
                if (val > bv[TOPK - 1]) {
#pragma unroll
                    for (int t = 0; t < TOPK; ++t) {
                        if (val > bv[t]) {
                            float fv = bv[t]; bv[t] = val; val = fv;
                            int   fi = bi[t]; bi[t] = id;  id  = fi;
                        }
                    }
                }
            }
        }
        const int q = blockIdx.x * QSUPER + tid;
        float* ov = cand_val + ((size_t)q * NCHUNK + blockIdx.y) * TOPK;
        int*   oi = cand_idx + ((size_t)q * NCHUNK + blockIdx.y) * TOPK;
#pragma unroll
        for (int t = 0; t < TOPK; ++t) { ov[t] = bv[t]; oi[t] = bi[t]; }
    }
}

// ---------------------------------------------------------------------------
// Kernel 2: reduce NCHUNK*5 candidates per query to final top-5, gather rows.
//   grid = B_Q blocks x 64 threads.
// ---------------------------------------------------------------------------
__global__ __launch_bounds__(64)
void reduce_gather_kernel(const float* __restrict__ cand_val,
                          const int*   __restrict__ cand_idx,
                          const float* __restrict__ memory,   // f32 [K][64]
                          float* __restrict__ out) {          // [B][5][64]
    __shared__ int sidx[TOPK];
    const int q = blockIdx.x;

    if (threadIdx.x == 0) {
        float bv[TOPK]; int bi[TOPK];
#pragma unroll
        for (int t = 0; t < TOPK; ++t) { bv[t] = -3.0e38f; bi[t] = 0; }
        const float* cv = cand_val + (size_t)q * NCHUNK * TOPK;
        const int*   ci = cand_idx + (size_t)q * NCHUNK * TOPK;
        for (int e = 0; e < NCHUNK * TOPK; ++e) {
            float val = cv[e];
            if (val > bv[TOPK - 1]) {
                int id = ci[e];
#pragma unroll
                for (int t = 0; t < TOPK; ++t) {
                    if (val > bv[t]) {
                        float fv = bv[t]; bv[t] = val; val = fv;
                        int   fi = bi[t]; bi[t] = id;  id  = fi;
                    }
                }
            }
        }
#pragma unroll
        for (int t = 0; t < TOPK; ++t) sidx[t] = bi[t];
    }
    __syncthreads();

    for (int e = threadIdx.x; e < TOPK * D_DIM; e += 64) {
        int t = e >> 6, d = e & 63;
        int id = sidx[t]; id = id < 0 ? 0 : id;
        out[(size_t)q * TOPK * D_DIM + e] = memory[(size_t)id * D_DIM + d];
    }
}

// ---------------------------------------------------------------------------
// Host-side launcher
// ---------------------------------------------------------------------------
extern "C" void kernel_launch(void* const* d_in, const int* in_sizes, int n_in,
                              void* d_out, int out_size, void* d_ws, size_t ws_size,
                              hipStream_t stream) {
    const float* query  = (const float*)d_in[0];   // [1024, 64] f32
    const float* memory = (const float*)d_in[1];   // [262144, 64] f32
    // d_in[2] = topk (scalar, == 5) — compile-time constant here.
    float* out = (float*)d_out;

    // workspace layout (bytes):
    //   [0, 33554432)            memory as bf16
    //   [33554432, 33685504)     queries as bf16
    //   [33685504, 34996224)     candidate values  [B][NCHUNK][5] f32
    //   [34996224, 36306944)     candidate indices [B][NCHUNK][5] i32
    unsigned char* ws = (unsigned char*)d_ws;
    unsigned short* Mbf = (unsigned short*)ws;
    unsigned short* Qbf = (unsigned short*)(ws + (size_t)K_MEM * D_DIM * 2);
    float* cand_val = (float*)(ws + 33685504ull);
    int*   cand_idx = (int*)(ws + 34996224ull);

    // 0) f32 -> bf16 conversion passes
    {
        int n4m = (K_MEM * D_DIM) / 4;                 // 4,194,304
        cvt_bf16_kernel<<<n4m / 256, 256, 0, stream>>>(memory, Mbf, n4m);
        int n4q = (B_Q * D_DIM) / 4;                   // 16,384
        cvt_bf16_kernel<<<n4q / 256, 256, 0, stream>>>(query, Qbf, n4q);
    }

    // 1) WMMA scoring + hierarchical top-5
    {
        dim3 grid(NQT, NCHUNK);                        // 16 x 64 blocks
        score_topk_kernel<<<grid, 128, 0, stream>>>(Mbf, Qbf, cand_val, cand_idx);
    }

    // 2) final reduction + row gather (f32 rows from original bank)
    reduce_gather_kernel<<<B_Q, 64, 0, stream>>>(cand_val, cand_idx, memory, out);
}
